// Fp8Linear_27135603376323
// MI455X (gfx1250) — 8-bit matrix operand order repaired and checked by reading, compiled, not run on hardware
//
#include <hip/hip_runtime.h>
#include <cstdint>
#include <cstddef>


typedef int   v16i __attribute__((ext_vector_type(16)));
typedef float v8f  __attribute__((ext_vector_type(8)));

#define LDS_STRIDE 144
#define TILE_BYTES (128 * LDS_STRIDE)

__device__ __forceinline__ unsigned int f32_to_e4m3(float f) {
    unsigned int u    = __float_as_uint(f);
    unsigned int sign = (u >> 24) & 0x80u;
    float a = fabsf(f);
    if (!(a == a)) return sign | 0x7fu;
    if (a > 448.0f) a = 448.0f;
    unsigned int bits = __float_as_uint(a);
    int e = (int)(bits >> 23) - 127;
    if (e < -10) return sign;
    unsigned int m = (bits & 0x7fffffu) | 0x800000u;
    int shift = (e < -6) ? (20 + (-6 - e)) : 20;
    unsigned int q     = m >> shift;
    unsigned int rem   = m & ((1u << shift) - 1u);
    unsigned int halfw = 1u << (shift - 1);
    q += (rem > halfw || (rem == halfw && (q & 1u))) ? 1u : 0u;
    unsigned int res;
    if (e < -6) {
        res = q;
    } else {
        if (q >= 16u) { q >>= 1; ++e; }
        res = ((unsigned int)(e + 7) << 3) | (q & 7u);
        if (res > 0x7eu) res = 0x7eu;
    }
    return sign | res;
}

__global__ void init_ws_kernel(unsigned int* amax) {
    amax[0] = 0u;
    amax[1] = 0u;
}

__global__ void amax_kernel(const float4* __restrict__ x, int n4,
                            unsigned int* __restrict__ amax) {
    float m = 0.0f;
    for (int i = blockIdx.x * blockDim.x + threadIdx.x; i < n4;
         i += gridDim.x * blockDim.x) {
        float4 v = x[i];
        m = fmaxf(m, fmaxf(fmaxf(fabsf(v.x), fabsf(v.y)),
                           fmaxf(fabsf(v.z), fabsf(v.w))));
    }
    #pragma unroll
    for (int off = 16; off > 0; off >>= 1)
        m = fmaxf(m, __shfl_xor(m, off));
    if ((threadIdx.x & 31) == 0)
        atomicMax(amax, __float_as_uint(m));
}

__global__ void quant_kernel(const float4* __restrict__ x, int n4,
                             const unsigned int* __restrict__ amax,
                             unsigned int* __restrict__ q) {
    float a = fmaxf(__uint_as_float(*amax), 1e-12f);
    float s = 448.0f / a;
    for (int i = blockIdx.x * blockDim.x + threadIdx.x; i < n4;
         i += gridDim.x * blockDim.x) {
        float4 v = x[i];
        unsigned int r =  f32_to_e4m3(v.x * s)
                       | (f32_to_e4m3(v.y * s) << 8)
                       | (f32_to_e4m3(v.z * s) << 16)
                       | (f32_to_e4m3(v.w * s) << 24);
        q[i] = r;
    }
}

__device__ __forceinline__ void async_copy16(unsigned int lds_off,
                                             const unsigned char* g) {
    asm volatile("global_load_async_to_lds_b128 %0, %1, off"
                 :: "v"(lds_off), "v"((unsigned long long)(uintptr_t)g)
                 : "memory");
}

__device__ __forceinline__ void wait_asynccnt0() {
    asm volatile("s_wait_asynccnt 0x0" ::: "memory");
}

__global__ __launch_bounds__(256)
void gemm_fp8_kernel(const unsigned char* __restrict__ qA,
                     const unsigned char* __restrict__ qW,
                     const float* __restrict__ bias,
                     const unsigned int* __restrict__ amax,
                     float* __restrict__ out,
                     int M, int N, int K) {
    __shared__ __align__(16) unsigned char sBuf[4 * TILE_BYTES];

    const int t     = threadIdx.x;
    const int lane  = t & 31;
    const int wid   = t >> 5;
    const int h     = lane >> 4;
    const int l16   = lane & 15;
    const int waveM = wid & 1;
    const int waveN = wid >> 1;

    const int blkN = blockIdx.x * 128;
    const int blkM = blockIdx.y * 128;

    v8f acc[4][2];
    #pragma unroll
    for (int mi = 0; mi < 4; ++mi)
        #pragma unroll
        for (int ni = 0; ni < 2; ++ni)
            #pragma unroll
            for (int e = 0; e < 8; ++e)
                acc[mi][ni][e] = 0.0f;

    const int loadRow   = t >> 3;
    const int loadChunk = (t & 7) * 16;

    const unsigned int ldsBase = (unsigned int)(uintptr_t)(&sBuf[0]);
    const unsigned int ldsA0   = ldsBase + loadRow * LDS_STRIDE + loadChunk;
    const unsigned int ldsB0   = ldsA0 + TILE_BYTES;
    const unsigned char* gA0 = qA + (size_t)(blkM + loadRow) * K + loadChunk;
    const unsigned char* gB0 = qW + (size_t)(blkN + loadRow) * K + loadChunk;

    const int nk = K >> 7;

    #pragma unroll
    for (int p = 0; p < 4; ++p) {
        async_copy16(ldsA0 + p * 32 * LDS_STRIDE, gA0 + (size_t)p * 32 * K);
        async_copy16(ldsB0 + p * 32 * LDS_STRIDE, gB0 + (size_t)p * 32 * K);
    }

    for (int i = 0; i < nk; ++i) {
        wait_asynccnt0();
        __syncthreads();

        if (i + 1 < nk) {
            const unsigned int bufOff = (unsigned int)(((i + 1) & 1) * 2 * TILE_BYTES);
            const size_t kOff = (size_t)(i + 1) << 7;
            #pragma unroll
            for (int p = 0; p < 4; ++p) {
                async_copy16(ldsA0 + bufOff + p * 32 * LDS_STRIDE,
                             gA0 + (size_t)p * 32 * K + kOff);
                async_copy16(ldsB0 + bufOff + p * 32 * LDS_STRIDE,
                             gB0 + (size_t)p * 32 * K + kOff);
            }
        }

        const unsigned char* ldsA = sBuf + (i & 1) * 2 * TILE_BYTES;
        const unsigned char* ldsB = ldsA + TILE_BYTES;

        v16i bf[2];
        #pragma unroll
        for (int ni = 0; ni < 2; ++ni) {
            const unsigned char* bp =
                ldsB + (size_t)(waveN * 32 + ni * 16 + l16) * LDS_STRIDE + h * 16;
            #pragma unroll
            for (int g = 0; g < 4; ++g) {
                int4 d = *(const int4*)(bp + g * 32);
                bf[ni][4 * g + 0] = d.x;
                bf[ni][4 * g + 1] = d.y;
                bf[ni][4 * g + 2] = d.z;
                bf[ni][4 * g + 3] = d.w;
            }
        }

        #pragma unroll
        for (int mi = 0; mi < 4; ++mi) {
            v16i af;
            const unsigned char* ap =
                ldsA + (size_t)(waveM * 64 + mi * 16 + l16) * LDS_STRIDE + h * 16;
            #pragma unroll
            for (int half = 0; half < 2; ++half) {
                #pragma unroll
                for (int k4 = 0; k4 < 4; ++k4) {
                    int2 d = *(const int2*)(ap + half * 64 + (k4 >> 1) * 32 + (k4 & 1) * 8);
                    af[half * 8 + 2 * k4]     = d.x;
                    af[half * 8 + 2 * k4 + 1] = d.y;
                }
            }
            #pragma unroll
            for (int ni = 0; ni < 2; ++ni) {
                acc[mi][ni] = __builtin_amdgcn_wmma_f32_16x16x128_fp8_fp8(
                    af, bf[ni], (short)0, acc[mi][ni], false, false);
            }
        }
    }

    const float aIn = fmaxf(__uint_as_float(amax[0]), 1e-12f);
    const float aW  = fmaxf(__uint_as_float(amax[1]), 1e-12f);
    const float scale = (aIn * (1.0f / 448.0f)) * (aW * (1.0f / 448.0f));

    #pragma unroll
    for (int mi = 0; mi < 4; ++mi) {
        const int m0 = blkM + waveM * 64 + mi * 16 + h * 8;
        #pragma unroll
        for (int ni = 0; ni < 2; ++ni) {
            const int n0 = blkN + waveN * 32 + ni * 16 + l16;
            const float bv = bias[n0];
            #pragma unroll
            for (int v = 0; v < 8; ++v) {
                out[(size_t)(m0 + v) * N + n0] = acc[mi][ni][v] * scale + bv;
            }
        }
    }
}

extern "C" void kernel_launch(void* const* d_in, const int* in_sizes, int n_in,
                              void* d_out, int out_size, void* d_ws, size_t ws_size,
                              hipStream_t stream) {
    const float* inp  = (const float*)d_in[0];
    const float* wgt  = (const float*)d_in[1];
    const float* bias = (const float*)d_in[2];
    float* out = (float*)d_out;

    const int N = in_sizes[2];
    const int K = in_sizes[1] / N;
    const int M = in_sizes[0] / K;

    unsigned char* ws   = (unsigned char*)d_ws;
    unsigned int*  amax = (unsigned int*)ws;
    unsigned char* qA   = ws + 256;
    unsigned char* qW   = qA + (size_t)M * K;

    const int nIn4 = (M * K) / 4;
    const int nW4  = (N * K) / 4;

    init_ws_kernel<<<1, 1, 0, stream>>>(amax);

    int blkA = (nIn4 + 255) / 256; if (blkA > 4096) blkA = 4096;
    int blkW = (nW4  + 255) / 256; if (blkW > 4096) blkW = 4096;
    amax_kernel<<<blkA, 256, 0, stream>>>((const float4*)inp, nIn4, amax + 0);
    amax_kernel<<<blkW, 256, 0, stream>>>((const float4*)wgt, nW4,  amax + 1);

    int qblkA = (nIn4 + 255) / 256; if (qblkA > 8192) qblkA = 8192;
    int qblkW = (nW4  + 255) / 256; if (qblkW > 8192) qblkW = 8192;
    quant_kernel<<<qblkA, 256, 0, stream>>>((const float4*)inp, nIn4, amax + 0,
                                            (unsigned int*)qA);
    quant_kernel<<<qblkW, 256, 0, stream>>>((const float4*)wgt, nW4, amax + 1,
                                            (unsigned int*)qW);

    dim3 grid(N / 128, M / 128);
    gemm_fp8_kernel<<<grid, 256, 0, stream>>>(qA, qW, bias, amax, out, M, N, K);
}
